// GaussianGAT_59184649339058
// MI455X (gfx1250) — matrix operand order repaired and checked by reading, not compiled, not run on hardware
//
#include <hip/hip_runtime.h>

typedef __attribute__((ext_vector_type(16))) _Float16 v16h;
typedef __attribute__((ext_vector_type(8)))  float    v8f;

#define HDIM   64
#define INDIM  7
#define NGRAPH 256
#define FCDIM  128
#define NCLS   2
#define SLOPE  0.2f
#define ENC_NEG_INF 0x007FFFFFu

// ---- ordered-uint encoding so atomicMax(uint) == max over floats ----
static __device__ __forceinline__ unsigned int encf(float f) {
  unsigned int u = __float_as_uint(f);
  return (u & 0x80000000u) ? ~u : (u | 0x80000000u);
}
static __device__ __forceinline__ float decf(unsigned int k) {
  unsigned int u = (k & 0x80000000u) ? (k ^ 0x80000000u) : ~k;
  return __uint_as_float(u);
}

// ---- layer-0 transform: K=7, scalar MACs (too small for WMMA) ----
__global__ __launch_bounds__(256)
void k_transform_in(const float* __restrict__ x,
                    const float* __restrict__ Wl, const float* __restrict__ bl,
                    const float* __restrict__ Wr, const float* __restrict__ br,
                    float* __restrict__ xlr, int N) {
  int i = blockIdx.x * blockDim.x + threadIdx.x;
  if (i >= N * HDIM) return;
  int sel = blockIdx.y;
  const float* W = sel ? Wr : Wl;
  const float* b = sel ? br : bl;
  float* o = xlr + (size_t)sel * (size_t)N * HDIM;
  int n = i >> 6, j = i & 63;
  float a = b[j];
#pragma unroll
  for (int k = 0; k < INDIM; ++k) a += x[n * INDIM + k] * W[j * INDIM + k];
  o[i] = a;
}

// ---- f32 -> f16 conversions ----
__global__ __launch_bounds__(256)
void k_f2h(const float* __restrict__ s, _Float16* __restrict__ d, int n) {
  int i = blockIdx.x * blockDim.x + threadIdx.x;
  if (i < n) d[i] = (_Float16)s[i];
}
__global__ __launch_bounds__(256)
void k_wcvt(const float* __restrict__ Wl, const float* __restrict__ Wr,
            _Float16* __restrict__ w16) {
  int i = blockIdx.x * blockDim.x + threadIdx.x;
  if (i < 2 * HDIM * HDIM)
    w16[i] = (_Float16)(i < HDIM * HDIM ? Wl[i] : Wr[i - HDIM * HDIM]);
}

// ---- hidden transform: y[N,64] = h[N,64] @ W[64,64]^T + b via v_wmma_f32_16x16x32_f16
// One wave = one 16-row node tile; 4 j-tiles x 2 k-steps = 8 WMMAs per wave.
__global__ __launch_bounds__(128)
void k_wmma_transform(const _Float16* __restrict__ h16,
                      const _Float16* __restrict__ w16,   // [2][64*64], l then r
                      const float* __restrict__ bl, const float* __restrict__ br,
                      float* __restrict__ xlr,            // [2][N*64]
                      int nrows, int ntiles) {
  __shared__ _Float16 As[4][16 * HDIM];
  const int sel = blockIdx.y;
  const _Float16* W  = w16 + sel * (HDIM * HDIM);
  const float* bias  = sel ? br : bl;
  float* out         = xlr + (size_t)sel * (size_t)nrows * HDIM;
  const int wv   = threadIdx.x >> 5;
  const int lane = threadIdx.x & 31;
  const int tile = blockIdx.x * 4 + wv;
  if (tile >= ntiles) return;
  const int base = tile * 16;

  // cooperative per-wave copy of the 16x64 f16 A tile (2 KB) into LDS
  {
    const uint4* src = (const uint4*)(h16 + (size_t)base * HDIM);
    uint4* dst = (uint4*)As[wv];
#pragma unroll
    for (int i = 0; i < 4; ++i) dst[i * 32 + lane] = src[i * 32 + lane];
  }

  // A fragment: lane L -> row M=L%16, group g=L/16,
  // VGPR v holds K = g*16 + v*2 , +1 (contiguous, matches B)
  const int m = lane & 15, g = lane >> 4;
  const unsigned int* row32 = (const unsigned int*)&As[wv][m * HDIM];
  union AU { unsigned int u[8]; v16h h; } A0, A1;
#pragma unroll
  for (int v = 0; v < 8; ++v) {
    int k = g * 16 + v * 2;
    A0.u[v] = row32[k >> 1];
    A1.u[v] = row32[(k + 32) >> 1];
  }

#pragma unroll
  for (int jt = 0; jt < 4; ++jt) {
    const int jbase = jt * 16;
    const int n = lane & 15;
    // B fragment: lane (n,g) holds K = kk + g*16 .. +15 of column jbase+n,
    // i.e. 16 contiguous halves of w16 row jbase+n  (B = W^T)
    union AU B0, B1;
    const unsigned int* w0 = (const unsigned int*)(W + (size_t)(jbase + n) * HDIM + g * 16);
    const unsigned int* w1 = (const unsigned int*)(W + (size_t)(jbase + n) * HDIM + 32 + g * 16);
#pragma unroll
    for (int v = 0; v < 8; ++v) { B0.u[v] = w0[v]; B1.u[v] = w1[v]; }

    v8f acc = {};
    acc = __builtin_amdgcn_wmma_f32_16x16x32_f16(false, A0.h, false, B0.h,
                                                 (short)0, acc, false, false);
    acc = __builtin_amdgcn_wmma_f32_16x16x32_f16(false, A1.h, false, B1.h,
                                                 (short)0, acc, false, false);
    const float bv = bias[jbase + n];
#pragma unroll
    for (int r = 0; r < 8; ++r) {
      int Mrow = g * 8 + r;                       // D layout: M = (lane/16)*8 + r, N = lane%16
      out[(size_t)(base + Mrow) * HDIM + jbase + n] = acc[r] + bv;
    }
  }
}

// ---- per-layer segment state init: zero aggregation target, denom, -inf max keys ----
__global__ __launch_bounds__(256)
void k_seg_init(float* __restrict__ h, float* __restrict__ denom,
                unsigned int* __restrict__ mkey, int N) {
  int i = blockIdx.x * blockDim.x + threadIdx.x;
  if (i < N * HDIM) h[i] = 0.f;
  if (i < N) { denom[i] = 0.f; mkey[i] = ENC_NEG_INF; }
}

// ---- edge pass 1: e = leakyrelu(xl[src]+xr[dst]) . att ; segment max ----
__global__ __launch_bounds__(256)
void k_edge_score(const float* __restrict__ xl, const float* __restrict__ xr,
                  const float* __restrict__ att,
                  const int* __restrict__ src, const int* __restrict__ dst,
                  float* __restrict__ e, unsigned int* __restrict__ mkey, int E) {
  const int wid  = (blockIdx.x * blockDim.x + threadIdx.x) >> 5;
  const int lane = threadIdx.x & 31;
  if (wid >= E) return;
  const int s = src[wid], d = dst[wid];
  const float2 l = *(const float2*)(xl + (size_t)s * HDIM + lane * 2);
  const float2 r = *(const float2*)(xr + (size_t)d * HDIM + lane * 2);
  const float2 a = *(const float2*)(att + lane * 2);
  float v0 = l.x + r.x; v0 = (v0 > 0.f) ? v0 : SLOPE * v0;
  float v1 = l.y + r.y; v1 = (v1 > 0.f) ? v1 : SLOPE * v1;
  float sum = v0 * a.x + v1 * a.y;
#pragma unroll
  for (int off = 16; off >= 1; off >>= 1) sum += __shfl_xor(sum, off, 32);
  if (lane == 0) { e[wid] = sum; atomicMax(mkey + d, encf(sum)); }
}

// ---- edge pass 2: w = exp(e - m[dst]); denom[dst] += w ----
__global__ __launch_bounds__(256)
void k_edge_exp(const int* __restrict__ dst, const unsigned int* __restrict__ mkey,
                float* __restrict__ e, float* __restrict__ denom, int E) {
  int i = blockIdx.x * blockDim.x + threadIdx.x;
  if (i >= E) return;
  int d = dst[i];
  float w = expf(e[i] - decf(mkey[d]));
  e[i] = w;
  atomicAdd(denom + d, w);
}

// ---- edge pass 3: h[dst] += (w/denom[dst]) * xl[src] ----
__global__ __launch_bounds__(256)
void k_edge_agg(const float* __restrict__ xl,
                const int* __restrict__ src, const int* __restrict__ dst,
                const float* __restrict__ w, const float* __restrict__ denom,
                float* __restrict__ h, int E) {
  const int wid  = (blockIdx.x * blockDim.x + threadIdx.x) >> 5;
  const int lane = threadIdx.x & 31;
  if (wid >= E) return;
  const int s = src[wid], d = dst[wid];
  const float alpha = w[wid] / denom[d];
  const float2 l = *(const float2*)(xl + (size_t)s * HDIM + lane * 2);
  atomicAdd(h + (size_t)d * HDIM + lane * 2 + 0, alpha * l.x);
  atomicAdd(h + (size_t)d * HDIM + lane * 2 + 1, alpha * l.y);
}

// ---- h = elu(h + bias) ----
__global__ __launch_bounds__(256)
void k_bias_elu(float* __restrict__ h, const float* __restrict__ bias, int n64) {
  int i = blockIdx.x * blockDim.x + threadIdx.x;
  if (i >= n64) return;
  float v = h[i] + bias[i & 63];
  h[i] = (v > 0.f) ? v : expm1f(v);
}

// ---- pooling ----
__global__ __launch_bounds__(256)
void k_pool_init(float* __restrict__ zg, float* __restrict__ cnts) {
  int i = blockIdx.x * blockDim.x + threadIdx.x;
  if (i < NGRAPH * HDIM) zg[i] = 0.f;
  if (i < NGRAPH) cnts[i] = 0.f;
}
__global__ __launch_bounds__(256)
void k_reparam_pool(const float* __restrict__ hm, const float* __restrict__ hv,
                    const float* __restrict__ eps, const int* __restrict__ batch,
                    float* __restrict__ zg, float* __restrict__ cnts, int N) {
  int i = blockIdx.x * blockDim.x + threadIdx.x;
  if (i >= N * HDIM) return;
  int n = i >> 6, hd = i & 63;
  float z = hm[i] + eps[i] * expf(0.5f * hv[i]);
  int g = batch[n];
  atomicAdd(zg + g * HDIM + hd, z);
  if (hd == 0) atomicAdd(cnts + g, 1.f);
}

// ---- MLP head + log_softmax (one block per graph) ----
__global__ __launch_bounds__(128)
void k_head(const float* __restrict__ zg, const float* __restrict__ cnts,
            const float* __restrict__ fc1W, const float* __restrict__ fc1b,
            const float* __restrict__ fc2W, const float* __restrict__ fc2b,
            float* __restrict__ out) {
  __shared__ float zgm[HDIM];
  __shared__ float h1[FCDIM];
  __shared__ float lg[NCLS];
  int g = blockIdx.x, t = threadIdx.x;
  if (t < HDIM) zgm[t] = zg[g * HDIM + t] / cnts[g];
  __syncthreads();
  float a = fc1b[t];
#pragma unroll 8
  for (int k = 0; k < HDIM; ++k) a += zgm[k] * fc1W[t * HDIM + k];
  h1[t] = fmaxf(a, 0.f);
  __syncthreads();
  if (t < NCLS) {
    float l = fc2b[t];
    for (int k = 0; k < FCDIM; ++k) l += h1[k] * fc2W[t * FCDIM + k];
    lg[t] = l;
  }
  __syncthreads();
  if (t < NCLS) {
    float mx = fmaxf(lg[0], lg[1]);
    float lse = mx + logf(expf(lg[0] - mx) + expf(lg[1] - mx));
    out[g * NCLS + t] = lg[t] - lse;
  }
}

extern "C" void kernel_launch(void* const* d_in, const int* in_sizes, int n_in,
                              void* d_out, int out_size, void* d_ws, size_t ws_size,
                              hipStream_t stream) {
  (void)n_in; (void)out_size; (void)ws_size;
  const float* x    = (const float*)d_in[0];
  const float* eps  = (const float*)d_in[1];
  const int*   edge = (const int*)d_in[2];
  const int*   batch= (const int*)d_in[3];
  const int N  = in_sizes[0] / INDIM;     // 100000
  const int E  = in_sizes[2] / 2;         // 1700000
  const int* srcI = edge;
  const int* dstI = edge + E;
  const int N64 = N * HDIM;

  float* out = (float*)d_out;
  float* h_m = out + NGRAPH * NCLS;       // mean lives in d_out
  float* h_v = h_m + N64;                 // log_var lives in d_out

  // workspace carve (≈72 MB)
  char* p = (char*)d_ws;
  auto carve = [&](size_t bytes) -> char* {
    char* r = p; p += (bytes + 255) & ~(size_t)255; return r;
  };
  float*        xlr   = (float*)carve(2 * (size_t)N64 * sizeof(float));
  _Float16*     h16   = (_Float16*)carve((size_t)N64 * sizeof(_Float16));
  _Float16*     w16   = (_Float16*)carve(2 * HDIM * HDIM * sizeof(_Float16));
  float*        ebuf  = (float*)carve((size_t)E * sizeof(float));
  unsigned int* mkey  = (unsigned int*)carve((size_t)N * sizeof(unsigned int));
  float*        denom = (float*)carve((size_t)N * sizeof(float));
  float*        zg    = (float*)carve((size_t)NGRAPH * HDIM * sizeof(float));
  float*        cnts  = (float*)carve((size_t)NGRAPH * sizeof(float));

  const int TB    = 256;
  const int gN64  = (N64 + TB - 1) / TB;
  const int gE    = (E + TB - 1) / TB;
  const int gEw   = (int)(((size_t)E * 32 + TB - 1) / TB);   // wave-per-edge grids
  const int ntiles= (N + 15) / 16;
  const int gWMMA = (ntiles + 3) / 4;

  for (int brn = 0; brn < 2; ++brn) {
    float* h = brn ? h_v : h_m;
    const int base0 = 4 + brn * 6;    // m0_* / v0_*
    const int baseh = 16 + brn * 6;   // m_* / v_*

    // ---- layer 0 (IN=7 -> 64) ----
    k_transform_in<<<dim3(gN64, 2), TB, 0, stream>>>(
        x, (const float*)d_in[base0 + 0], (const float*)d_in[base0 + 1],
        (const float*)d_in[base0 + 2], (const float*)d_in[base0 + 3], xlr, N);
    k_seg_init<<<gN64, TB, 0, stream>>>(h, denom, mkey, N);
    k_edge_score<<<gEw, TB, 0, stream>>>(xlr, xlr + N64,
        (const float*)d_in[base0 + 4], srcI, dstI, ebuf, mkey, E);
    k_edge_exp<<<gE, TB, 0, stream>>>(dstI, mkey, ebuf, denom, E);
    k_edge_agg<<<gEw, TB, 0, stream>>>(xlr, srcI, dstI, ebuf, denom, h, E);
    k_bias_elu<<<gN64, TB, 0, stream>>>(h, (const float*)d_in[base0 + 5], N64);

    // ---- layers 1..2 (64 -> 64, WMMA) ----
    for (int li = 0; li < 2; ++li) {
      const float* Wl = (const float*)d_in[baseh + 0] + li * HDIM * HDIM;
      const float* bl = (const float*)d_in[baseh + 1] + li * HDIM;
      const float* Wr = (const float*)d_in[baseh + 2] + li * HDIM * HDIM;
      const float* br = (const float*)d_in[baseh + 3] + li * HDIM;
      const float* at = (const float*)d_in[baseh + 4] + li * HDIM;
      const float* bs = (const float*)d_in[baseh + 5] + li * HDIM;

      k_f2h<<<gN64, TB, 0, stream>>>(h, h16, N64);
      k_wcvt<<<(2 * HDIM * HDIM + TB - 1) / TB, TB, 0, stream>>>(Wl, Wr, w16);
      k_wmma_transform<<<dim3(gWMMA, 2), 128, 0, stream>>>(h16, w16, bl, br, xlr, N, ntiles);
      k_seg_init<<<gN64, TB, 0, stream>>>(h, denom, mkey, N);
      k_edge_score<<<gEw, TB, 0, stream>>>(xlr, xlr + N64, at, srcI, dstI, ebuf, mkey, E);
      k_edge_exp<<<gE, TB, 0, stream>>>(dstI, mkey, ebuf, denom, E);
      k_edge_agg<<<gEw, TB, 0, stream>>>(xlr, srcI, dstI, ebuf, denom, h, E);
      k_bias_elu<<<gN64, TB, 0, stream>>>(h, bs, N64);
    }
  }

  // ---- reparameterize + global mean pool + head ----
  k_pool_init<<<(NGRAPH * HDIM + TB - 1) / TB, TB, 0, stream>>>(zg, cnts);
  k_reparam_pool<<<gN64, TB, 0, stream>>>(h_m, h_v, eps, batch, zg, cnts, N);
  k_head<<<NGRAPH, 128, 0, stream>>>(zg, cnts,
      (const float*)d_in[28], (const float*)d_in[29],
      (const float*)d_in[30], (const float*)d_in[31], out);
}
